// DecomposableAttentionEntailment_85650237817378
// MI455X (gfx1250) — matrix operand order repaired and checked by reading, not compiled, not run on hardware
//
#include <hip/hip_runtime.h>
#include <hip/hip_bf16.h>

typedef __bf16 bf16x8  __attribute__((ext_vector_type(8)));
typedef __bf16 bf16x16 __attribute__((ext_vector_type(16)));
typedef float  v8f     __attribute__((ext_vector_type(8)));
typedef int    i32x4   __attribute__((vector_size(16)));   // matches builtin param type

#define Bsz 128
#define Lsz 512
#define Dsz 512
#define HWsz 512

// CDNA5 async global->LDS staging (ASYNCcnt path), with safe fallback.
#if defined(__gfx1250__) && __has_builtin(__builtin_amdgcn_global_load_async_to_lds_b128)
#define USE_ASYNC_LDS 1
#else
#define USE_ASYNC_LDS 0
#endif

__device__ __forceinline__ void async_copy32(const __bf16* gsrc, __bf16* lds_dst)
{
#if USE_ASYNC_LDS
    // b128 = 16 bytes; immediate offset applies to both global and LDS address.
    __builtin_amdgcn_global_load_async_to_lds_b128(
        (__attribute__((address_space(1))) i32x4*)gsrc,
        (__attribute__((address_space(3))) i32x4*)lds_dst, 0, 0);
    __builtin_amdgcn_global_load_async_to_lds_b128(
        (__attribute__((address_space(1))) i32x4*)gsrc,
        (__attribute__((address_space(3))) i32x4*)lds_dst, 16, 0);
#else
    *(bf16x16*)lds_dst = *(const bf16x16*)gsrc;
#endif
}

__device__ __forceinline__ void wait_async_then_barrier()
{
#if USE_ASYNC_LDS
#if __has_builtin(__builtin_amdgcn_s_wait_asynccnt)
    __builtin_amdgcn_s_wait_asynccnt(0);
#else
    asm volatile("s_wait_asynccnt 0x0" ::: "memory");
#endif
#endif
    __syncthreads();
}

// ---------------------------------------------------------------------------
// Generic bf16 WMMA GEMM: C = [relu](A @ B [^T]), f32 accumulate.
// Block tile 128x64, 8 waves (4 along M, 2 along N), each wave 32x32 (2x2 wmma).
// A: [M,K] bf16 row-major (lda). B: !TRANSB -> [K,N] (ldb); TRANSB -> [N,K] (ldb).
// Requires M%128==0, K%32==0. GUARDN=false additionally requires N%64==0.
// ---------------------------------------------------------------------------
template <bool RELU, bool F32OUT, bool TRANSB, bool GUARDN>
__global__ __launch_bounds__(256)
void gemm_wmma(const __bf16* __restrict__ A, long lda, long strideA,
               const __bf16* __restrict__ Bm, long ldb, long strideB,
               void* __restrict__ Cv, long ldc, long strideC,
               int M, int N, int K)
{
    constexpr int BM = 128, BN = 64, BK = 32;
    __shared__ __align__(32) __bf16 As[BM * BK];   // [m][k]
    __shared__ __align__(32) __bf16 Bs[BN * BK];   // [n][k]

    const int tid  = threadIdx.x;
    const int wave = tid >> 5;
    const int lane = tid & 31;
    const int wm   = (wave & 3) * 32;   // wave M offset in block tile
    const int wn   = (wave >> 2) * 32;  // wave N offset in block tile
    const int m0   = blockIdx.x * BM;
    const int n0   = blockIdx.y * BN;
    const int z    = blockIdx.z;

    A  += (long)z * strideA;
    Bm += (long)z * strideB;
    float*  Cf = (float*)Cv + (long)z * strideC;
    __bf16* Cb = (__bf16*)Cv + (long)z * strideC;

    // staging coordinates (hoisted)
    const int ar = tid >> 1;             // A row in tile
    const int ac = (tid & 1) * 16;       // A col segment
    const __bf16* aSrc = A + (long)(m0 + ar) * lda + ac;
    __bf16* aDst = &As[ar * BK + ac];

    v8f acc[2][2] = {};

    for (int k0 = 0; k0 < K; k0 += BK) {
        // ---- stage A tile: 128x32 bf16, 32B per thread (async -> LDS) ----
        async_copy32(aSrc + k0, aDst);
        if (k0 + BK < K) __builtin_prefetch(aSrc + k0 + BK, 0, 1);  // global_prefetch_b8

        // ---- stage B tile as Bs[n][k] ----
        if (TRANSB) {
            if (tid < 128) {
                const int n = tid >> 1;
                const int c = (tid & 1) * 16;
                async_copy32(Bm + (long)(n0 + n) * ldb + (k0 + c), &Bs[n * BK + c]);
            }
        } else {
            const int k    = tid >> 3;         // 0..31
            const int nseg = (tid & 7) * 8;    // 0..56
            const __bf16* src = Bm + (long)(k0 + k) * ldb + n0 + nseg;
            if (GUARDN) {
#pragma unroll
                for (int j = 0; j < 8; ++j) {
                    const int gn = n0 + nseg + j;
                    Bs[(nseg + j) * BK + k] = (gn < N) ? src[j] : (__bf16)0.0f;
                }
            } else {
                bf16x8 v = *(const bf16x8*)src;      // one 16B vector load
#pragma unroll
                for (int j = 0; j < 8; ++j)
                    Bs[(nseg + j) * BK + k] = v[j];  // transpose scatter (ds_store_b16)
            }
        }
        wait_async_then_barrier();

        // ---- wave-level WMMA: fragments in contiguous-K order (same for A and B) ----
        const int am = lane & 15;
        const int kb = (lane >> 4) * 16;   // A: 16 contiguous K
        const int bn = lane & 15;
        const int bk = (lane >> 4) * 16;   // B: 16 contiguous K
#pragma unroll
        for (int mi = 0; mi < 2; ++mi) {
            const int row = wm + mi * 16 + am;
            bf16x16 afrag = *(const bf16x16*)&As[row * BK + kb];
#pragma unroll
            for (int ni = 0; ni < 2; ++ni) {
                const int col = wn + ni * 16 + bn;
                bf16x16 bfrag = *(const bf16x16*)&Bs[col * BK + bk];
                acc[mi][ni] = __builtin_amdgcn_wmma_f32_16x16x32_bf16(
                    false, afrag, false, bfrag, (short)0, acc[mi][ni], false, false);
            }
        }
        __syncthreads();
    }

    // ---- store C (layout: VGPR i -> row (lane>>4)*8+i, col lane&15) ----
    const int cn    = lane & 15;
    const int rbase = (lane >> 4) * 8;
#pragma unroll
    for (int mi = 0; mi < 2; ++mi)
#pragma unroll
        for (int ni = 0; ni < 2; ++ni)
#pragma unroll
            for (int i = 0; i < 8; ++i) {
                const int gm = m0 + wm + mi * 16 + rbase + i;
                const int gn = n0 + wn + ni * 16 + cn;
                if (!GUARDN || gn < N) {
                    float v = acc[mi][ni][i];
                    if (RELU) v = v > 0.f ? v : 0.f;
                    if (F32OUT) Cf[(long)gm * ldc + gn] = v;
                    else        Cb[(long)gm * ldc + gn] = (__bf16)v;
                }
            }
}

// ---------------------------------------------------------------------------
// Elementwise helpers
// ---------------------------------------------------------------------------
__global__ void f2bf_kernel(const float* __restrict__ in, __bf16* __restrict__ out, int n)
{
    int i = blockIdx.x * blockDim.x + threadIdx.x;
    if (i < n) out[i] = (__bf16)in[i];
}

// masked embedding: out[row, d] (row stride ldo) = bf16(x[row*512+d] * mask[row])
__global__ void embed_mask_kernel(const float* __restrict__ x, const int* __restrict__ mask,
                                  __bf16* __restrict__ out, long ldo)
{
    long idx = (long)blockIdx.x * blockDim.x + threadIdx.x;   // over B*L*D
    long row = idx >> 9;
    int  d   = (int)(idx & 511);
    float m  = (float)mask[row];
    out[row * ldo + d] = (__bf16)(x[idx] * m);
}

// Flattened masked softmax over one batch's 512x512 f32 score matrix.
// Writes p2h bf16 and its transpose h2p bf16 (global max/sum are shared).
__global__ __launch_bounds__(256)
void flat_masked_softmax(const float* __restrict__ att,
                         const int* __restrict__ pmask, const int* __restrict__ hmask,
                         __bf16* __restrict__ p2h, __bf16* __restrict__ h2p)
{
    const int b   = blockIdx.x;
    const int tid = threadIdx.x;
    const float* a = att + (long)b * (Lsz * Lsz);
    const int* pm = pmask + b * Lsz;
    const int* hm = hmask + b * Lsz;
    __shared__ float red[256];

    // pass 1: masked max
    float mx = -3.4e38f;
    for (int i = tid; i < Lsz * Lsz; i += 256) {
        int p = i >> 9, h = i & 511;
        float v = (pm[p] & hm[h]) ? a[i] : -1e9f;
        mx = fmaxf(mx, v);
    }
    red[tid] = mx; __syncthreads();
    for (int s = 128; s > 0; s >>= 1) {
        if (tid < s) red[tid] = fmaxf(red[tid], red[tid + s]);
        __syncthreads();
    }
    mx = red[0]; __syncthreads();

    // pass 2: sum of exp
    float sum = 0.f;
    for (int i = tid; i < Lsz * Lsz; i += 256) {
        int p = i >> 9, h = i & 511;
        float v = (pm[p] & hm[h]) ? a[i] : -1e9f;
        sum += __expf(v - mx);
    }
    red[tid] = sum; __syncthreads();
    for (int s = 128; s > 0; s >>= 1) {
        if (tid < s) red[tid] += red[tid + s];
        __syncthreads();
    }
    const float inv = 1.f / red[0]; __syncthreads();

    // pass 3: write probs + transpose
    for (int i = tid; i < Lsz * Lsz; i += 256) {
        int p = i >> 9, h = i & 511;
        float v = (pm[p] & hm[h]) ? a[i] : -1e9f;
        float w = __expf(v - mx) * inv;          // masked -> underflows to 0
        p2h[(long)b * (Lsz * Lsz) + i]           = (__bf16)w;
        h2p[(long)b * (Lsz * Lsz) + h * Lsz + p] = (__bf16)w;
    }
}

// Sum compared_{p,h} over sequence axis -> agg bf16 [B, 1024]
__global__ void seq_sum_kernel(const __bf16* __restrict__ cp, const __bf16* __restrict__ ch,
                               __bf16* __restrict__ agg)
{
    int idx = blockIdx.x * blockDim.x + threadIdx.x;   // B*1024
    int b = idx >> 10, n = idx & 1023;
    const __bf16* src = (n < HWsz) ? (cp + (long)b * Lsz * HWsz + n)
                                   : (ch + (long)b * Lsz * HWsz + (n - HWsz));
    float s = 0.f;
    for (int l = 0; l < Lsz; ++l) s += (float)src[(long)l * HWsz];
    agg[idx] = (__bf16)s;
}

__global__ void softmax2_kernel(const float* __restrict__ logits, float* __restrict__ out)
{
    int b = blockIdx.x * blockDim.x + threadIdx.x;
    if (b < Bsz) {
        float a = logits[b * 2], c = logits[b * 2 + 1];
        float mx = fmaxf(a, c);
        float ea = __expf(a - mx), ec = __expf(c - mx);
        float inv = 1.f / (ea + ec);
        out[b * 2]     = ea * inv;
        out[b * 2 + 1] = ec * inv;
    }
}

// ---------------------------------------------------------------------------
extern "C" void kernel_launch(void* const* d_in, const int* in_sizes, int n_in,
                              void* d_out, int out_size, void* d_ws, size_t ws_size,
                              hipStream_t stream)
{
    (void)in_sizes; (void)n_in; (void)out_size; (void)ws_size;

    const float* premise    = (const float*)d_in[0];
    const float* hypothesis = (const float*)d_in[1];
    const int*   pmask      = (const int*)d_in[2];
    const int*   hmask      = (const int*)d_in[3];
    const float* attend_w0  = (const float*)d_in[4];
    const float* attend_w1  = (const float*)d_in[5];
    const float* compare_w0 = (const float*)d_in[6];
    const float* compare_w1 = (const float*)d_in[7];
    const float* agg_w0     = (const float*)d_in[8];
    const float* agg_w1     = (const float*)d_in[9];
    const float* scorer     = (const float*)d_in[10];
    float* out = (float*)d_out;

    const long M  = (long)Bsz * Lsz;      // 65536
    char* ws = (char*)d_ws;
    size_t off = 0;
    auto alloc = [&](size_t bytes) { size_t o = off; off = (off + bytes + 255) & ~(size_t)255; return o; };

    __bf16* catP  = (__bf16*)(ws + alloc(M * 1024 * 2));   // [M,1024]: left=p_emb, right=p2h_align
    __bf16* catH  = (__bf16*)(ws + alloc(M * 1024 * 2));
    __bf16* tmp0  = (__bf16*)(ws + alloc(M * 512 * 2));
    __bf16* projP = (__bf16*)(ws + alloc(M * 512 * 2));    // reused as compared_p
    __bf16* projH = (__bf16*)(ws + alloc(M * 512 * 2));    // reused as compared_h
    float*  att   = (float*)(ws + alloc(M * 512 * 4));
    __bf16* p2h   = (__bf16*)(ws + alloc(M * 512 * 2));
    __bf16* h2p   = (__bf16*)(ws + alloc(M * 512 * 2));
    __bf16* agg   = (__bf16*)(ws + alloc((size_t)Bsz * 1024 * 2));
    __bf16* h0b   = (__bf16*)(ws + alloc((size_t)Bsz * 512 * 2));
    __bf16* h1b   = (__bf16*)(ws + alloc((size_t)Bsz * 512 * 2));
    float*  logit = (float*)(ws + alloc((size_t)Bsz * 2 * 4));
    __bf16* aw0 = (__bf16*)(ws + alloc(512 * 512 * 2));
    __bf16* aw1 = (__bf16*)(ws + alloc(512 * 512 * 2));
    __bf16* cw0 = (__bf16*)(ws + alloc(1024 * 512 * 2));
    __bf16* cw1 = (__bf16*)(ws + alloc(512 * 512 * 2));
    __bf16* gw0 = (__bf16*)(ws + alloc(1024 * 512 * 2));
    __bf16* gw1 = (__bf16*)(ws + alloc(512 * 512 * 2));
    __bf16* scb = (__bf16*)(ws + alloc(512 * 2 * 2));

    auto cvt = [&](const float* src, __bf16* dst, int n) {
        f2bf_kernel<<<(n + 255) / 256, 256, 0, stream>>>(src, dst, n);
    };
    cvt(attend_w0, aw0, 512 * 512);   cvt(attend_w1, aw1, 512 * 512);
    cvt(compare_w0, cw0, 1024 * 512); cvt(compare_w1, cw1, 512 * 512);
    cvt(agg_w0, gw0, 1024 * 512);     cvt(agg_w1, gw1, 512 * 512);
    cvt(scorer, scb, 512 * 2);

    // masked embeddings -> left halves of cat buffers (row stride 1024)
    {
        long n = M * 512;
        embed_mask_kernel<<<(unsigned)(n / 256), 256, 0, stream>>>(premise, pmask, catP, 1024);
        embed_mask_kernel<<<(unsigned)(n / 256), 256, 0, stream>>>(hypothesis, hmask, catH, 1024);
    }

    dim3 gBig((unsigned)(M / 128), 8, 1);           // M=65536, N=512
    dim3 gBat(4, 8, Bsz);                           // batched 512x512x512
    const long bs = (long)Lsz * Lsz;                // 262144

    // Attend FF: relu(relu(emb @ aw0) @ aw1)
    gemm_wmma<true, false, false, false><<<gBig, 256, 0, stream>>>(catP, 1024, 0, aw0, 512, 0, tmp0, 512, 0, (int)M, 512, 512);
    gemm_wmma<true, false, false, false><<<gBig, 256, 0, stream>>>(tmp0, 512, 0, aw1, 512, 0, projP, 512, 0, (int)M, 512, 512);
    gemm_wmma<true, false, false, false><<<gBig, 256, 0, stream>>>(catH, 1024, 0, aw0, 512, 0, tmp0, 512, 0, (int)M, 512, 512);
    gemm_wmma<true, false, false, false><<<gBig, 256, 0, stream>>>(tmp0, 512, 0, aw1, 512, 0, projH, 512, 0, (int)M, 512, 512);

    // Pair scores: att[b] = projP[b] @ projH[b]^T  (f32 out)
    gemm_wmma<false, true, true, false><<<gBat, 256, 0, stream>>>(projP, 512, bs, projH, 512, bs, att, 512, bs, Lsz, Lsz, 512);

    // Flattened masked softmax (+ free transpose for h2p)
    flat_masked_softmax<<<Bsz, 256, 0, stream>>>(att, pmask, hmask, p2h, h2p);

    // Alignments -> right halves of cat buffers
    gemm_wmma<false, false, false, false><<<gBat, 256, 0, stream>>>(p2h, 512, bs, catH, 1024, (long)Lsz * 1024, catP + 512, 1024, (long)Lsz * 1024, Lsz, 512, 512);
    gemm_wmma<false, false, false, false><<<gBat, 256, 0, stream>>>(h2p, 512, bs, catP, 1024, (long)Lsz * 1024, catH + 512, 1024, (long)Lsz * 1024, Lsz, 512, 512);

    // Compare FF (K=1024 over the concat), reuse projP/projH as compared_{p,h}
    gemm_wmma<true, false, false, false><<<gBig, 256, 0, stream>>>(catP, 1024, 0, cw0, 512, 0, tmp0, 512, 0, (int)M, 512, 1024);
    gemm_wmma<true, false, false, false><<<gBig, 256, 0, stream>>>(tmp0, 512, 0, cw1, 512, 0, projP, 512, 0, (int)M, 512, 512);
    gemm_wmma<true, false, false, false><<<gBig, 256, 0, stream>>>(catH, 1024, 0, cw0, 512, 0, tmp0, 512, 0, (int)M, 512, 1024);
    gemm_wmma<true, false, false, false><<<gBig, 256, 0, stream>>>(tmp0, 512, 0, cw1, 512, 0, projH, 512, 0, (int)M, 512, 512);

    // Aggregate: sum over L, concat -> [B,1024]
    seq_sum_kernel<<<(Bsz * 1024) / 256, 256, 0, stream>>>(projP, projH, agg);

    // Aggregate FF + scorer
    dim3 gAgg(1, 8, 1);
    gemm_wmma<true, false, false, false><<<gAgg, 256, 0, stream>>>(agg, 1024, 0, gw0, 512, 0, h0b, 512, 0, Bsz, 512, 1024);
    gemm_wmma<true, false, false, false><<<gAgg, 256, 0, stream>>>(h0b, 512, 0, gw1, 512, 0, h1b, 512, 0, Bsz, 512, 512);
    dim3 gSc(1, 1, 1);
    gemm_wmma<false, true, false, true><<<gSc, 256, 0, stream>>>(h1b, 512, 0, scb, 2, 0, logit, 2, 0, Bsz, 2, 512);

    softmax2_kernel<<<1, 128, 0, stream>>>(logit, out);
}
